// EncoderAttention_9088150798449
// MI455X (gfx1250) — matrix operand order repaired and checked by reading, not compiled, not run on hardware
//
#include <hip/hip_runtime.h>
#include <hip/hip_bf16.h>

// ---------------------------------------------------------------------------
// EncoderAttention for MI455X (gfx1250): bf16 WMMA pipeline, v3
//   - all kernels 2-stage software pipelined (double-buffered LDS)
//   - branch-free staging, batched global_load_b128
//   K1: qkv = x @ w_qkv   (f32 -> bf16; Q row-major, K row-major, V transposed)
//   K2: cache rows [64,320) -> bf16 KV bufs (V transposed)
//   K3: flash attention (online softmax, WMMA bf16, LDS-staged K/V^T)
//   K4: out = attn_out @ w_proj + b
// ---------------------------------------------------------------------------

typedef __attribute__((ext_vector_type(16))) __bf16 v16bf;
typedef __attribute__((ext_vector_type(8)))  float  v8f;

#define NUM_HEADS 16
#define PREFIX_   64

constexpr int B_ = 64, N_ = 320, C_ = 1024, H_ = NUM_HEADS, D_ = 64;
constexpr int MAX_KV = 1024;
constexpr int CUR = 256;                       // cur_kv_len (fixed by setup)
constexpr int APP = N_ - PREFIX_;              // 256
constexpr int AVAIL = PREFIX_ + CUR + APP;     // 576
constexpr int BN = B_ * N_;                    // 20480
constexpr size_t QELEMS  = (size_t)B_ * H_ * N_ * D_;
constexpr size_t KVELEMS = (size_t)B_ * H_ * AVAIL * D_;

// Load a 16x32 A-fragment / 32x16 B-fragment (two contiguous 16B segments).
static __device__ __forceinline__ v16bf ld_frag(const __bf16* p0, const __bf16* p1) {
    union { v16bf v; uint4 u[2]; } f;
    f.u[0] = *(const uint4*)p0;
    f.u[1] = *(const uint4*)p1;
    return f.v;
}

static __device__ __forceinline__ v8f wmma_bf16(v16bf a, v16bf b, v8f c) {
    return __builtin_amdgcn_wmma_f32_16x16x32_bf16(
        false, a, false, b, (short)0, c, false, false);
}

// ---------------------------------------------------------------------------
// K1: QKV GEMM. M=BN, K=1024, Nout=3072. 128x64 tile/WG, 4 waves (32x64 each).
// ---------------------------------------------------------------------------
__global__ __launch_bounds__(128)
void qkv_gemm_kernel(const float* __restrict__ x, const float* __restrict__ w,
                     __bf16* __restrict__ qb, __bf16* __restrict__ kb,
                     __bf16* __restrict__ vbT)
{
    __shared__ __bf16 aT[2][128 * 32];  // [m][k]
    __shared__ __bf16 bT[2][64 * 32];   // [n][k]

    const int tid  = threadIdx.x;
    const int lane = tid & 31, wave = tid >> 5;
    const int hi   = lane >> 4, lo = lane & 15;
    const int m0   = blockIdx.x * 128;
    const int n0   = blockIdx.y * 64;

    // Uniform decomposition of the output-column tile (scalar branches).
    const int which = n0 >> 10;            // 0=q 1=k 2=v
    const int h     = (n0 & 1023) >> 6;
    const int bbase = m0 / N_;             // one scalar division
    const int nbase = m0 - bbase * N_;

    const int ar = tid >> 3, ak = (tid & 7) * 4;     // A staging coords
    const int bk = (tid >> 4) * 4, bn = (tid & 15) * 4; // B staging coords

    float4 fa[8];
    float4 fb[4];

    auto load_tiles = [&](int k0) {
        #pragma unroll
        for (int p = 0; p < 8; ++p)
            fa[p] = *(const float4*)&x[(size_t)(m0 + p * 16 + ar) * C_ + k0 + ak];
        #pragma unroll
        for (int i = 0; i < 4; ++i)
            fb[i] = *(const float4*)&w[(size_t)(k0 + bk + i) * 3072 + n0 + bn];
    };
    auto store_tiles = [&](int buf) {
        #pragma unroll
        for (int p = 0; p < 8; ++p) {
            __bf16 t4[4] = {(__bf16)fa[p].x, (__bf16)fa[p].y,
                            (__bf16)fa[p].z, (__bf16)fa[p].w};
            *(uint2*)&aT[buf][(p * 16 + ar) * 32 + ak] = *(uint2*)&t4[0];
        }
        float fv[4][4];
        #pragma unroll
        for (int i = 0; i < 4; ++i) {
            fv[i][0] = fb[i].x; fv[i][1] = fb[i].y;
            fv[i][2] = fb[i].z; fv[i][3] = fb[i].w;
        }
        #pragma unroll
        for (int j = 0; j < 4; ++j) {      // 4x4 register transpose
            __bf16 t4[4] = {(__bf16)fv[0][j], (__bf16)fv[1][j],
                            (__bf16)fv[2][j], (__bf16)fv[3][j]};
            *(uint2*)&bT[buf][(bn + j) * 32 + bk] = *(uint2*)&t4[0];
        }
    };

    v8f acc[2][4] = {};
    load_tiles(0);
    int buf = 0;
    for (int k0 = 0; k0 < C_; k0 += 32) {
        store_tiles(buf);
        __syncthreads();
        load_tiles((k0 + 32 < C_) ? (k0 + 32) : k0);   // next tile (clamped)

        v16bf bfr[4];
        #pragma unroll
        for (int nc = 0; nc < 4; ++nc) {
            const __bf16* brow = &bT[buf][(nc * 16 + lo) * 32 + hi * 16];
            bfr[nc] = ld_frag(brow, brow + 8);
        }
        #pragma unroll
        for (int a2 = 0; a2 < 2; ++a2) {
            const __bf16* arow = &aT[buf][(wave * 32 + a2 * 16 + lo) * 32];
            v16bf af = ld_frag(arow + hi * 16, arow + hi * 16 + 8);
            #pragma unroll
            for (int nc = 0; nc < 4; ++nc)
                acc[a2][nc] = wmma_bf16(af, bfr[nc], acc[a2][nc]);
        }
        buf ^= 1;
    }

    // Epilogue: scatter bf16 into attention layouts (no per-lane divisions).
    #pragma unroll
    for (int a2 = 0; a2 < 2; ++a2) {
        #pragma unroll
        for (int nc = 0; nc < 4; ++nc) {
            #pragma unroll
            for (int r = 0; r < 8; ++r) {
                const int rowoff = wave * 32 + a2 * 16 + r + 8 * hi;
                int n = nbase + rowoff;
                const int cross = (n >= N_);
                const int b = bbase + cross;
                n -= cross ? N_ : 0;
                const int d = nc * 16 + lo;
                const __bf16 val = (__bf16)acc[a2][nc][r];
                if (which == 0) {
                    qb[(((size_t)b * H_ + h) * N_ + n) * D_ + d] = val;
                } else {
                    const int pos = (n < PREFIX_) ? n : (n + CUR);
                    if (which == 1)
                        kb[(((size_t)b * H_ + h) * AVAIL + pos) * D_ + d] = val;
                    else  // V stored transposed: [bh][d][pos]
                        vbT[(((size_t)b * H_ + h) * D_ + d) * AVAIL + pos] = val;
                }
            }
        }
    }
}

// ---------------------------------------------------------------------------
// K2: cache rows [PREFIX, PREFIX+CUR) -> bf16 (K row-major, V transposed)
// ---------------------------------------------------------------------------
__global__ __launch_bounds__(256)
void cache_cvt_kernel(const float* __restrict__ kc, const float* __restrict__ vc,
                      __bf16* __restrict__ kb, __bf16* __restrict__ vbT)
{
    const size_t i = (size_t)blockIdx.x * blockDim.x + threadIdx.x; // 4 elems each
    const size_t total = (size_t)B_ * H_ * CUR * D_ / 4;
    if (i >= total) return;
    const size_t e   = i * 4;
    const int    d   = (int)(e % D_);
    const size_t t   = e / D_;
    const int    pos = (int)(t % CUR) + PREFIX_;
    const size_t bh  = t / CUR;
    const size_t src = (bh * MAX_KV + pos) * D_ + d;

    float4 fk = *(const float4*)&kc[src];
    float4 fv = *(const float4*)&vc[src];
    __bf16 tk[4] = {(__bf16)fk.x, (__bf16)fk.y, (__bf16)fk.z, (__bf16)fk.w};
    *(uint2*)&kb[(bh * AVAIL + pos) * D_ + d] = *(uint2*)&tk[0];
    __bf16 tv[4] = {(__bf16)fv.x, (__bf16)fv.y, (__bf16)fv.z, (__bf16)fv.w};
    #pragma unroll
    for (int j = 0; j < 4; ++j)
        vbT[(bh * D_ + d + j) * AVAIL + pos] = tv[j];
}

// ---------------------------------------------------------------------------
// K3: flash attention. One (b,h) per WG; 4 waves, each owns a 16-row Q tile.
//     Double-buffered K/V^T chunks; all WMMA fragments contiguous b128 pairs.
// ---------------------------------------------------------------------------
__global__ __launch_bounds__(128)
void attn_kernel(const __bf16* __restrict__ qb, const __bf16* __restrict__ kb,
                 const __bf16* __restrict__ vbT, __bf16* __restrict__ ob)
{
    __shared__ __bf16 kt[2][32 * 64];    // K chunk  [m][d]
    __shared__ __bf16 vtT[2][64 * 32];   // V^T chunk [d][m]
    __shared__ __bf16 pb[4][16 * 32];    // per-wave P tile [n][m]

    const int tid  = threadIdx.x;
    const int lane = tid & 31, wave = tid >> 5;
    const int hi   = lane >> 4, lo = lane & 15;
    const int bh   = blockIdx.y;
    const int n0   = (blockIdx.x * 4 + wave) * 16;
    const size_t qbase  = (size_t)bh * N_ * D_;
    const size_t kvbase = (size_t)bh * AVAIL * D_;
    const size_t vtbase = (size_t)bh * D_ * AVAIL;
    const float  scale  = 0.125f;        // D^-0.5

    const __bf16* qrow = &qb[qbase + (size_t)(n0 + lo) * D_];
    const v16bf qf0 = ld_frag(qrow + hi * 16,      qrow + hi * 16 + 8);
    const v16bf qf1 = ld_frag(qrow + 32 + hi * 16, qrow + 32 + hi * 16 + 8);

    const int krw = tid >> 2, kcb = (tid & 3) * 16;  // K staging coords
    const int vd  = tid >> 1, vmb = (tid & 1) * 16;  // V^T staging coords
    uint4 rk0, rk1, rv0, rv1;

    auto load_kv = [&](int m0c) {
        const size_t g  = kvbase + (size_t)(m0c + krw) * D_ + kcb;
        rk0 = *(const uint4*)&kb[g];
        rk1 = *(const uint4*)&kb[g + 8];
        const size_t gv = vtbase + (size_t)vd * AVAIL + m0c + vmb;
        rv0 = *(const uint4*)&vbT[gv];
        rv1 = *(const uint4*)&vbT[gv + 8];
        __builtin_prefetch(&kb[g + (size_t)32 * D_], 0, 3);   // global_prefetch_b8
        __builtin_prefetch(&vbT[gv + 32], 0, 3);
    };
    auto store_kv = [&](int buf) {
        *(uint4*)&kt[buf][krw * 64 + kcb]     = rk0;
        *(uint4*)&kt[buf][krw * 64 + kcb + 8] = rk1;
        *(uint4*)&vtT[buf][vd * 32 + vmb]     = rv0;
        *(uint4*)&vtT[buf][vd * 32 + vmb + 8] = rv1;
    };

    float m_run[8], l_run[8];
    #pragma unroll
    for (int r = 0; r < 8; ++r) { m_run[r] = -3.0e38f; l_run[r] = 0.f; }
    v8f o[4] = {};

    load_kv(0);
    int buf = 0;
    for (int m0 = 0; m0 < AVAIL; m0 += 32) {
        store_kv(buf);
        __syncthreads();
        load_kv((m0 + 32 < AVAIL) ? (m0 + 32) : m0);

        // S = Q * K^T (two 16x16 tiles)
        v8f s[2];
        #pragma unroll
        for (int t = 0; t < 2; ++t) {
            const __bf16* krow = &kt[buf][(t * 16 + lo) * 64];
            v16bf b0 = ld_frag(krow + hi * 16,      krow + hi * 16 + 8);
            v16bf b1 = ld_frag(krow + 32 + hi * 16, krow + 32 + hi * 16 + 8);
            v8f z = {};
            z = wmma_bf16(qf0, b0, z);
            z = wmma_bf16(qf1, b1, z);
            s[t] = z;
        }

        // Online softmax (row = r + 8*hi, cols striped over 16 lanes)
        #pragma unroll
        for (int r = 0; r < 8; ++r) {
            float s0 = s[0][r] * scale;
            float s1 = s[1][r] * scale;
            float mx = fmaxf(s0, s1);
            #pragma unroll
            for (int off = 1; off < 16; off <<= 1)
                mx = fmaxf(mx, __shfl_xor(mx, off, 32));
            const float mnew  = fmaxf(m_run[r], mx);
            const float alpha = __expf(m_run[r] - mnew);
            m_run[r] = mnew;
            l_run[r] *= alpha;
            #pragma unroll
            for (int c = 0; c < 4; ++c) o[c][r] *= alpha;
            const float p0 = __expf(s0 - mnew);
            const float p1 = __expf(s1 - mnew);
            s[0][r] = p0; s[1][r] = p1;
            float rs = p0 + p1;
            #pragma unroll
            for (int off = 1; off < 16; off <<= 1)
                rs += __shfl_xor(rs, off, 32);
            l_run[r] += rs;
        }

        // P: D-layout regs -> A-layout via per-wave LDS round trip
        __bf16* myp = pb[wave];
        #pragma unroll
        for (int t = 0; t < 2; ++t)
            #pragma unroll
            for (int r = 0; r < 8; ++r)
                myp[(r + 8 * hi) * 32 + t * 16 + lo] = (__bf16)s[t][r];
        asm volatile("s_wait_dscnt 0" ::: "memory");

        const __bf16* prow = &myp[lo * 32];
        const v16bf pf = ld_frag(prow + hi * 16, prow + hi * 16 + 8);

        // O += P(16x32) * V(32x64); V^T rows give contiguous B-fragments
        #pragma unroll
        for (int c = 0; c < 4; ++c) {
            const __bf16* vrow = &vtT[buf][(c * 16 + lo) * 32 + hi * 16];
            v16bf bf = ld_frag(vrow, vrow + 8);
            o[c] = wmma_bf16(pf, bf, o[c]);
        }
        buf ^= 1;
    }

    // Finalize: O / l -> bf16 (B,N,C) with C = h*64 + d
    const int b = bh / H_, h = bh % H_;
    #pragma unroll
    for (int r = 0; r < 8; ++r) {
        const float inv = 1.0f / l_run[r];
        const int nrow = n0 + r + 8 * hi;
        #pragma unroll
        for (int c = 0; c < 4; ++c) {
            const int d = c * 16 + lo;
            ob[((size_t)b * N_ + nrow) * C_ + h * D_ + d] = (__bf16)(o[c][r] * inv);
        }
    }
}

// ---------------------------------------------------------------------------
// K4: projection GEMM. out(f32) = ob(bf16) @ w_proj + bias. 128x64 tile/WG.
// ---------------------------------------------------------------------------
__global__ __launch_bounds__(128)
void proj_gemm_kernel(const __bf16* __restrict__ a, const float* __restrict__ w,
                      const float* __restrict__ bias, float* __restrict__ out)
{
    __shared__ __bf16 aT[2][128 * 32];
    __shared__ __bf16 bT[2][64 * 32];

    const int tid  = threadIdx.x;
    const int lane = tid & 31, wave = tid >> 5;
    const int hi   = lane >> 4, lo = lane & 15;
    const int m0   = blockIdx.x * 128;
    const int n0   = blockIdx.y * 64;

    const int ar = tid >> 2, ak = (tid & 3) * 8;        // A staging coords
    const int bk = (tid >> 4) * 4, bn = (tid & 15) * 4; // B staging coords

    uint4  ua[4];
    float4 fb[4];

    auto load_tiles = [&](int k0) {
        #pragma unroll
        for (int p = 0; p < 4; ++p)
            ua[p] = *(const uint4*)&a[(size_t)(m0 + p * 32 + ar) * C_ + k0 + ak];
        #pragma unroll
        for (int i = 0; i < 4; ++i)
            fb[i] = *(const float4*)&w[(size_t)(k0 + bk + i) * C_ + n0 + bn];
    };
    auto store_tiles = [&](int buf) {
        #pragma unroll
        for (int p = 0; p < 4; ++p)
            *(uint4*)&aT[buf][(p * 32 + ar) * 32 + ak] = ua[p];
        float fv[4][4];
        #pragma unroll
        for (int i = 0; i < 4; ++i) {
            fv[i][0] = fb[i].x; fv[i][1] = fb[i].y;
            fv[i][2] = fb[i].z; fv[i][3] = fb[i].w;
        }
        #pragma unroll
        for (int j = 0; j < 4; ++j) {
            __bf16 t4[4] = {(__bf16)fv[0][j], (__bf16)fv[1][j],
                            (__bf16)fv[2][j], (__bf16)fv[3][j]};
            *(uint2*)&bT[buf][(bn + j) * 32 + bk] = *(uint2*)&t4[0];
        }
    };

    v8f acc[2][4] = {};
    load_tiles(0);
    int buf = 0;
    for (int k0 = 0; k0 < C_; k0 += 32) {
        store_tiles(buf);
        __syncthreads();
        load_tiles((k0 + 32 < C_) ? (k0 + 32) : k0);

        v16bf bfr[4];
        #pragma unroll
        for (int nc = 0; nc < 4; ++nc) {
            const __bf16* brow = &bT[buf][(nc * 16 + lo) * 32 + hi * 16];
            bfr[nc] = ld_frag(brow, brow + 8);
        }
        #pragma unroll
        for (int a2 = 0; a2 < 2; ++a2) {
            const __bf16* arow = &aT[buf][(wave * 32 + a2 * 16 + lo) * 32];
            v16bf af = ld_frag(arow + hi * 16, arow + hi * 16 + 8);
            #pragma unroll
            for (int nc = 0; nc < 4; ++nc)
                acc[a2][nc] = wmma_bf16(af, bfr[nc], acc[a2][nc]);
        }
        buf ^= 1;
    }

    #pragma unroll
    for (int a2 = 0; a2 < 2; ++a2) {
        #pragma unroll
        for (int nc = 0; nc < 4; ++nc) {
            #pragma unroll
            for (int r = 0; r < 8; ++r) {
                const int mg = m0 + wave * 32 + a2 * 16 + r + 8 * hi;
                const int ng = n0 + nc * 16 + lo;
                out[(size_t)mg * C_ + ng] = acc[a2][nc][r] + bias[ng];
            }
        }
    }
}

// ---------------------------------------------------------------------------
extern "C" void kernel_launch(void* const* d_in, const int* in_sizes, int n_in,
                              void* d_out, int out_size, void* d_ws, size_t ws_size,
                              hipStream_t stream)
{
    const float* x       = (const float*)d_in[0];
    const float* k_cache = (const float*)d_in[1];
    const float* v_cache = (const float*)d_in[2];
    const float* w_qkv   = (const float*)d_in[3];
    const float* w_proj  = (const float*)d_in[4];
    const float* b_proj  = (const float*)d_in[5];
    float* out = (float*)d_out;

    __bf16* qb  = (__bf16*)d_ws;
    __bf16* kb  = qb + QELEMS;
    __bf16* vbT = kb + KVELEMS;
    __bf16* ob  = vbT + KVELEMS;

    qkv_gemm_kernel<<<dim3(BN / 128, 3072 / 64), 128, 0, stream>>>(x, w_qkv, qb, kb, vbT);

    const int cvt_threads = (int)((size_t)B_ * H_ * CUR * D_ / 4);
    cache_cvt_kernel<<<dim3((cvt_threads + 255) / 256), 256, 0, stream>>>(k_cache, v_cache, kb, vbT);

    attn_kernel<<<dim3(N_ / 64, B_ * H_), 128, 0, stream>>>(qb, kb, vbT, ob);

    proj_gemm_kernel<<<dim3(BN / 128, C_ / 64), 128, 0, stream>>>(ob, w_proj, b_proj, out);
}
